// DepthTransformer_3667902071050
// MI455X (gfx1250) — matrix operand order repaired and checked by reading, not compiled, not run on hardware
//
#include <hip/hip_runtime.h>
#include <hip/hip_bf16.h>
#include <cstdint>

typedef __attribute__((ext_vector_type(16))) __bf16 v16bf;
typedef __attribute__((ext_vector_type(8)))  float  v8f;

#define BM 128
#define BN 64
#define BK 32
#define ASTRIDE 40   // halfwords per A row in LDS (padded, 80B = 16B-aligned rows)
#define BSTRIDE 36   // halfwords per B row in LDS (padded)

__device__ __forceinline__ unsigned int pack2_bf16(float lo, float hi) {
    unsigned int ul = __float_as_uint(lo);
    unsigned int uh = __float_as_uint(hi);
    ul = (ul + 0x7FFFu + ((ul >> 16) & 1u)) >> 16;                 // RNE
    uh = (uh + 0x7FFFu + ((uh >> 16) & 1u)) & 0xFFFF0000u;
    return ul | uh;
}
__device__ __forceinline__ unsigned short f32_to_bf16(float f) {
    unsigned int u = __float_as_uint(f);
    return (unsigned short)((u + 0x7FFFu + ((u >> 16) & 1u)) >> 16);
}

// ---------------------------------------------------------------------------
// Tiled WMMA GEMM: C[M,N] = A[M,K] * B[K,N] (+bias[m]) (+res), fp32 I/O,
// bf16 compute (v_wmma_f32_16x16x32_bf16). Workgroup tile 128x64, BK=32,
// 8 waves, each wave -> 32x32 outputs (4 WMMAs / K-step, A-frag reuse).
// flags: bit0 = A transposed (A(m,k)=A[k*lda+m]); bit1 = B is implicit im2col
//        of a (K/9, 32, 32) image, 3x3, pad 1 (requires N == 1024).
// Per-z: A += ((a_period>0)? z%a_period : z)*Az ; B += z*Bz ; C += z*Cz.
// ---------------------------------------------------------------------------
__global__ __launch_bounds__(256)
void wmma_gemm_bf16(const float* __restrict__ A, const float* __restrict__ B,
                    float* __restrict__ C, const float* __restrict__ bias,
                    const float* __restrict__ res,
                    int M, int N, int K, int lda, int ldb, int ldc,
                    long long Az, long long Bz, long long Cz,
                    int a_period, int flags)
{
    __shared__ __attribute__((aligned(16))) unsigned short As[BM * ASTRIDE];
    __shared__ __attribute__((aligned(16))) unsigned short Bs[BN * BSTRIDE];

    const long long z = blockIdx.z;
    A += ((a_period > 0) ? (z % a_period) : z) * Az;
    B += z * Bz;
    C += z * Cz;
    if (res) res += z * Cz;

    const int m0 = blockIdx.y * BM;
    const int n0 = blockIdx.x * BN;
    const int tid = threadIdx.x;
    const bool a_trans = (flags & 1) != 0;
    const bool im2col  = (flags & 2) != 0;

    const int wave = tid >> 5;
    const int lane = tid & 31;
    const int wm = (wave & 3) * 32;     // wave's row offset (two 16-row frags)
    const int wn = (wave >> 2) * 32;    // wave's col offset (two 16-col frags)

    // staging coordinates (fixed per thread)
    const int am  = tid >> 1;           // A row 0..127
    const int akq = (tid & 1) << 4;     // A k-offset 0 or 16 (16 elems/thread)
    const int bk  = tid >> 3;           // B k-row 0..31
    const int bnq = (tid & 7) << 3;     // B n-offset (8 elems/thread)

    v8f acc00 = {0,0,0,0,0,0,0,0}, acc01 = {0,0,0,0,0,0,0,0};
    v8f acc10 = {0,0,0,0,0,0,0,0}, acc11 = {0,0,0,0,0,0,0,0};

    const int ksteps = (K + BK - 1) / BK;
    for (int ks = 0; ks < ksteps; ++ks) {
        const int k0 = ks * BK;
        // ================= stage A (128 x 32) =================
        {
            const int gm  = m0 + am;
            const int gk0 = k0 + akq;
            unsigned int pk[8];
            if (!a_trans && gm < M && gk0 + 15 < K) {
                const float* src = A + (long long)gm * lda + gk0;
                #pragma unroll
                for (int q = 0; q < 4; ++q) {
                    float4 f = *(const float4*)(src + 4 * q);
                    pk[2*q]   = pack2_bf16(f.x, f.y);
                    pk[2*q+1] = pack2_bf16(f.z, f.w);
                }
                if (ks + 1 < ksteps && k0 + BK < K)
                    __builtin_prefetch(src + BK, 0, 0);
            } else {
                #pragma unroll
                for (int j = 0; j < 8; ++j) {
                    int ka = gk0 + 2*j, kb = ka + 1;
                    float a = 0.f, b = 0.f;
                    if (gm < M && ka < K)
                        a = a_trans ? A[(long long)ka * lda + gm]
                                    : A[(long long)gm * lda + ka];
                    if (gm < M && kb < K)
                        b = a_trans ? A[(long long)kb * lda + gm]
                                    : A[(long long)gm * lda + kb];
                    pk[j] = pack2_bf16(a, b);
                }
            }
            uint4* dst = (uint4*)&As[am * ASTRIDE + akq];
            dst[0] = make_uint4(pk[0], pk[1], pk[2], pk[3]);
            dst[1] = make_uint4(pk[4], pk[5], pk[6], pk[7]);
        }
        // ================= stage B (32 x 64) -> Bs[n][k] =================
        {
            const int gk  = k0 + bk;
            const int gn0 = n0 + bnq;
            float vals[8] = {0,0,0,0,0,0,0,0};
            if (gk < K) {
                if (im2col) {
                    // hoisted im2col decode: one k-row per thread
                    int c = gk / 9, r = gk - c * 9;
                    int dy = r / 3 - 1;
                    int dx = r - (r / 3) * 3 - 1;
                    int py = (gn0 >> 5) + dy;       // uniform across the 8
                    if (py >= 0 && py < 32) {
                        const float* img = B + (long long)c * 1024 + py * 32;
                        int pxb = (gn0 & 31) + dx;
                        #pragma unroll
                        for (int j = 0; j < 8; ++j) {
                            int px = pxb + j;
                            if ((unsigned)px < 32u) vals[j] = img[px];
                        }
                    }
                } else if (gn0 + 7 < N) {
                    const float* src = B + (long long)gk * ldb + gn0;
                    float4 f0 = *(const float4*)src;
                    float4 f1 = *(const float4*)(src + 4);
                    vals[0]=f0.x; vals[1]=f0.y; vals[2]=f0.z; vals[3]=f0.w;
                    vals[4]=f1.x; vals[5]=f1.y; vals[6]=f1.z; vals[7]=f1.w;
                    if (ks + 1 < ksteps && k0 + BK < K)
                        __builtin_prefetch(src + (long long)BK * ldb, 0, 0);
                } else {
                    #pragma unroll
                    for (int j = 0; j < 8; ++j)
                        if (gn0 + j < N) vals[j] = B[(long long)gk * ldb + gn0 + j];
                }
            }
            #pragma unroll
            for (int j = 0; j < 8; ++j)
                Bs[(bnq + j) * BSTRIDE + bk] = f32_to_bf16(vals[j]);
        }
        __syncthreads();

        // ========== fragments (ISA 7.12.2 bf16 wave32 layouts) ==========
        union { v16bf v; unsigned int u[8]; } fa0, fa1, fb0, fb1;
        {
            const int mA = wm + (lane & 15);
            const int halfA = (lane >> 4) * 8;
            const int nA = wn + (lane & 15);
            const int halfB = (lane >> 4) * 16;
            #pragma unroll
            for (int j = 0; j < 8; ++j) {
                int kA = ((j & 4) << 2) + ((j & 3) << 1) + halfA; // {0..7,16..23}+half
                int kB = ((j & 4) << 2) + ((j & 3) << 1) + halfA; // {0..7,16..23}+half
                fa0.u[j] = *(const unsigned int*)&As[mA * ASTRIDE + kA];
                fa1.u[j] = *(const unsigned int*)&As[(mA + 16) * ASTRIDE + kA];
                fb0.u[j] = *(const unsigned int*)&Bs[nA * BSTRIDE + kB];
                fb1.u[j] = *(const unsigned int*)&Bs[(nA + 16) * BSTRIDE + kB];
            }
        }
        acc00 = __builtin_amdgcn_wmma_f32_16x16x32_bf16(false, fa0.v, false, fb0.v,
                                                        (short)0, acc00, false, false);
        acc01 = __builtin_amdgcn_wmma_f32_16x16x32_bf16(false, fa0.v, false, fb1.v,
                                                        (short)0, acc01, false, false);
        acc10 = __builtin_amdgcn_wmma_f32_16x16x32_bf16(false, fa1.v, false, fb0.v,
                                                        (short)0, acc10, false, false);
        acc11 = __builtin_amdgcn_wmma_f32_16x16x32_bf16(false, fa1.v, false, fb1.v,
                                                        (short)0, acc11, false, false);
        __syncthreads();
    }

    // ---- store (C/D layout: VGPR r -> m = r + (lane>>4)*8 ; n = lane&15)
    const int half8 = (lane >> 4) * 8;
    const int nA = n0 + wn + (lane & 15);
    const int nB = nA + 16;
    #pragma unroll
    for (int r = 0; r < 8; ++r) {
        const int mA = m0 + wm + r + half8;
        const int mB = mA + 16;
        if (mA < M) {
            float bz = bias ? bias[mA] : 0.f;
            if (nA < N) {
                float v = acc00[r] + bz;
                if (res) v += res[(long long)mA * ldc + nA];
                C[(long long)mA * ldc + nA] = v;
            }
            if (nB < N) {
                float v = acc01[r] + bz;
                if (res) v += res[(long long)mA * ldc + nB];
                C[(long long)mA * ldc + nB] = v;
            }
        }
        if (mB < M) {
            float bz = bias ? bias[mB] : 0.f;
            if (nA < N) {
                float v = acc10[r] + bz;
                if (res) v += res[(long long)mB * ldc + nA];
                C[(long long)mB * ldc + nA] = v;
            }
            if (nB < N) {
                float v = acc11[r] + bz;
                if (res) v += res[(long long)mB * ldc + nB];
                C[(long long)mB * ldc + nB] = v;
            }
        }
    }
}

// ---------------------------------------------------------------------------
// GroupNorm + activation (+ optional depth pos-embed add).
// grid = (G, B). in/out are (B, C, S). mode: 0 none, 1 SiLU, 2 ReLU.
// If pos != null: out += pos[(s/pos_sp)*C + c].
// ---------------------------------------------------------------------------
__global__ __launch_bounds__(256)
void groupnorm_act(const float* __restrict__ in, float* __restrict__ out,
                   const float* __restrict__ gamma, const float* __restrict__ beta,
                   const float* __restrict__ pos,
                   int C, int S, int G, int mode, int pos_sp)
{
    const int g = blockIdx.x, b = blockIdx.y;
    const int cpg = C / G;
    const long long base = (long long)b * C * S + (long long)g * cpg * S;
    const float* src = in + base;
    float* dst = out + base;
    const long long n = (long long)cpg * S;

    float s = 0.f, s2 = 0.f;
    for (long long i = threadIdx.x; i < n; i += 256) {
        float v = src[i];
        s += v; s2 += v * v;
    }
    __shared__ float red0[256], red1[256];
    red0[threadIdx.x] = s; red1[threadIdx.x] = s2;
    __syncthreads();
    for (int off = 128; off > 0; off >>= 1) {
        if (threadIdx.x < off) {
            red0[threadIdx.x] += red0[threadIdx.x + off];
            red1[threadIdx.x] += red1[threadIdx.x + off];
        }
        __syncthreads();
    }
    const float mean = red0[0] / (float)n;
    const float var  = red1[0] / (float)n - mean * mean;
    const float rstd = rsqrtf(var + 1e-5f);

    for (long long i = threadIdx.x; i < n; i += 256) {
        int c  = g * cpg + (int)(i / S);
        int sp = (int)(i % S);
        float v = (src[i] - mean) * rstd * gamma[c] + beta[c];
        if (mode == 1)      v = v / (1.f + __expf(-v));
        else if (mode == 2) v = v > 0.f ? v : 0.f;
        if (pos) v += pos[(sp / pos_sp) * C + c];
        dst[i] = v;
    }
}

// ---------------------------------------------------------------------------
// Attention core: per pixel, wave = head (wave32), lane = depth:
//   sim[d] = scale * sum_e qt[h][e]*ctx[e][d];  attn = softmax_d(sim)
//   ctxa[h][e] = sum_d attn[d]*ctx[e][d]
// ctx: (B,64,32,1024)   qt/ctxa: (B,8,64,1024)
// ---------------------------------------------------------------------------
__global__ __launch_bounds__(256)
void depth_attn_core(const float* __restrict__ ctx, const float* __restrict__ qt,
                     float* __restrict__ ctxa, float scale)
{
    __shared__ float ctxL[64][33];
    __shared__ float qtL[8][64];
    const int pix = blockIdx.x;
    const int b   = blockIdx.y;
    const int tid = threadIdx.x;

    const float* cb = ctx + (long long)b * 64 * 32 * 1024;
    for (int i = tid; i < 64 * 32; i += 256) {
        int e = i >> 5, d = i & 31;
        ctxL[e][d] = cb[((long long)e * 32 + d) * 1024 + pix];
    }
    const float* qb = qt + (long long)b * 8 * 64 * 1024;
    for (int i = tid; i < 8 * 64; i += 256) {
        int h = i >> 6, e = i & 63;
        qtL[h][e] = qb[((long long)h * 64 + e) * 1024 + pix];
    }
    __syncthreads();

    const int head = tid >> 5;
    const int d    = tid & 31;
    float sim = 0.f;
    #pragma unroll 8
    for (int e = 0; e < 64; ++e) sim += qtL[head][e] * ctxL[e][d];
    sim *= scale;

    float mx = sim;
    for (int off = 16; off > 0; off >>= 1) mx = fmaxf(mx, __shfl_xor(mx, off, 32));
    float ex = __expf(sim - mx);
    float sum = ex;
    for (int off = 16; off > 0; off >>= 1) sum += __shfl_xor(sum, off, 32);
    const float attn = ex / sum;

    float a0 = 0.f, a1 = 0.f;
    #pragma unroll
    for (int dd = 0; dd < 32; ++dd) {
        float ad = __shfl(attn, dd, 32);
        a0 += ad * ctxL[d][dd];
        a1 += ad * ctxL[d + 32][dd];
    }
    float* ob = ctxa + ((long long)b * 8 + head) * 64 * 1024;
    ob[(long long)d * 1024 + pix]        = a0;
    ob[(long long)(d + 32) * 1024 + pix] = a1;
}

// ---------------------------------------------------------------------------
extern "C" void kernel_launch(void* const* d_in, const int* in_sizes, int n_in,
                              void* d_out, int out_size, void* d_ws, size_t ws_size,
                              hipStream_t stream) {
    (void)in_sizes; (void)n_in; (void)out_size; (void)ws_size;
    const float* x        = (const float*)d_in[0];   // (6,320,32,32)
    const float* context  = (const float*)d_in[1];   // (6,64,32,32,32)
    const float* w_in     = (const float*)d_in[2];
    const float* b_in     = (const float*)d_in[3];
    const float* gn_in_g  = (const float*)d_in[4];
    const float* gn_in_b  = (const float*)d_in[5];
    const float* w_ctx    = (const float*)d_in[6];
    const float* gn_ctx_g = (const float*)d_in[7];
    const float* gn_ctx_b = (const float*)d_in[8];
    const float* pos_emb  = (const float*)d_in[9];
    const float* wq       = (const float*)d_in[10];
    const float* wk       = (const float*)d_in[11];
    const float* wv       = (const float*)d_in[12];
    const float* w_ao     = (const float*)d_in[13];
    const float* gn1_g    = (const float*)d_in[14];
    const float* gn1_b    = (const float*)d_in[15];
    const float* conv1_w  = (const float*)d_in[16];
    const float* gn2_g    = (const float*)d_in[17];
    const float* gn2_b    = (const float*)d_in[18];
    const float* conv2_w  = (const float*)d_in[19];
    float* out = (float*)d_out;

    const int Bn = 6, HW = 1024, IN = 320, CD = 64, D = 32;
    const long long IMG = (long long)IN * HW;         // 327680
    const long long CTX = (long long)CD * D * HW;     // 2097152

    float* ws   = (float*)d_ws;
    float* xp   = ws;                                  // 6*320*1024
    float* ctx  = xp   + Bn * IMG;                     // 6*64*32*1024
    float* q    = ctx  + Bn * CTX;                     // 6*320*1024
    float* qt   = q    + Bn * IMG;                     // 6*8*64*1024
    float* ctxa = qt   + (long long)Bn * 8 * 64 * HW;  // 6*8*64*1024
    float* att  = ctxa + (long long)Bn * 8 * 64 * HW;  // 6*320*1024
    float* out2 = att  + Bn * IMG;                     // 6*320*1024
    float* y1   = out2 + Bn * IMG;                     // 6*320*1024

    dim3 blk(256);
    dim3 gBig((HW + BN - 1) / BN, (IN + BM - 1) / BM, Bn);     // 320x1024
    dim3 gCtx((D * HW + BN - 1) / BN, 1, Bn);                  // 64x32768
    dim3 gHead((HW + BN - 1) / BN, 1, Bn * 8);                 // per-head

    // 1) xp = w_in @ x + b_in
    wmma_gemm_bf16<<<gBig, blk, 0, stream>>>(w_in, x, xp, b_in, nullptr,
        IN, HW, IN, IN, HW, HW, 0, IMG, IMG, 0, 0);
    // 2) GN(8)+SiLU
    groupnorm_act<<<dim3(8, Bn), blk, 0, stream>>>(xp, xp, gn_in_g, gn_in_b,
        nullptr, IN, HW, 8, 1, 1);
    // 3) ctx = w_ctx @ context
    wmma_gemm_bf16<<<gCtx, blk, 0, stream>>>(w_ctx, context, ctx, nullptr, nullptr,
        CD, D * HW, CD, CD, D * HW, D * HW, 0, CTX, CTX, 0, 0);
    // 4) GN(8)+ReLU + pos_emb over depth
    groupnorm_act<<<dim3(8, Bn), blk, 0, stream>>>(ctx, ctx, gn_ctx_g, gn_ctx_b,
        pos_emb, CD, D * HW, 8, 2, HW);
    // 5) q = wq @ xp
    wmma_gemm_bf16<<<gBig, blk, 0, stream>>>(wq, xp, q, nullptr, nullptr,
        IN, HW, IN, IN, HW, HW, 0, IMG, IMG, 0, 0);
    // 6) qt[h] = wk_h^T @ q_h  (M=64, K=40, N=1024; z = b*8+h)
    wmma_gemm_bf16<<<gHead, blk, 0, stream>>>(wk, q, qt, nullptr, nullptr,
        64, HW, 40, CD, HW, HW, 40 * CD, 40 * HW, 64 * HW, 8, 1);
    // 7) softmax over depth, ctxa = ctx @ attn
    depth_attn_core<<<dim3(HW, Bn), blk, 0, stream>>>(ctx, qt, ctxa,
        0.15811388300841898f);
    // 8) att[h] = wv_h @ ctxa_h  (M=40, K=64, N=1024)
    wmma_gemm_bf16<<<gHead, blk, 0, stream>>>(wv, ctxa, att, nullptr, nullptr,
        40, HW, 64, CD, HW, HW, 40 * CD, 64 * HW, 40 * HW, 8, 0);
    // 9) out2 = w_attn_out @ att
    wmma_gemm_bf16<<<gBig, blk, 0, stream>>>(w_ao, att, out2, nullptr, nullptr,
        IN, HW, IN, IN, HW, HW, 0, IMG, IMG, 0, 0);
    // 10) GN(8)+ReLU (gn1)
    groupnorm_act<<<dim3(8, Bn), blk, 0, stream>>>(out2, out2, gn1_g, gn1_b,
        nullptr, IN, HW, 8, 2, 1);
    // 11) y1 = conv1 (implicit im2col GEMM, K = 320*9)
    wmma_gemm_bf16<<<gBig, blk, 0, stream>>>(conv1_w, out2, y1, nullptr, nullptr,
        IN, HW, IN * 9, IN * 9, HW, HW, 0, IMG, IMG, 0, 2);
    // 12) GN(8)+ReLU (gn2)
    groupnorm_act<<<dim3(8, Bn), blk, 0, stream>>>(y1, y1, gn2_g, gn2_b,
        nullptr, IN, HW, 8, 2, 1);
    // 13) out = conv2(y1) + x  (residual fused into store)
    wmma_gemm_bf16<<<gBig, blk, 0, stream>>>(conv2_w, y1, out, nullptr, x,
        IN, HW, IN * 9, IN * 9, HW, HW, 0, IMG, IMG, 0, 2);
}
